// REG_RNN_8693013807612
// MI455X (gfx1250) — matrix operand order repaired and checked by reading, not compiled, not run on hardware
//
#include <hip/hip_runtime.h>

typedef __attribute__((ext_vector_type(16))) _Float16 v16h;
typedef __attribute__((ext_vector_type(8)))  float    v8f;

#define B_TOT 256
#define T_LEN 4096
#define H_DIM 32

union V8FU  { float    f[8]; v8f  v; };
union V16HU { unsigned u[8]; v16h v; };

// Native v_tanh_f32 on gfx1250 (listed as a TRANS op in the CDNA5 ISA).
// Fallback: tanh(x) = 1 - 2/(2^(x*2*log2(e)) + 1) — one v_exp + one v_rcp.
__device__ __forceinline__ float fast_tanh(float x) {
#if __has_builtin(__builtin_amdgcn_tanhf)
  return __builtin_amdgcn_tanhf(x);
#elif __has_builtin(__builtin_amdgcn_exp2f)
  float e = __builtin_amdgcn_exp2f(x * 2.8853900817779268f); // 2/ln(2)
  return 1.0f - 2.0f * __builtin_amdgcn_rcpf(e + 1.0f);
#else
  float e = __expf(2.0f * x);
  return 1.0f - 2.0f * __builtin_amdgcn_rcpf(e + 1.0f);
#endif
}

// One wave (32 lanes) owns a 16-row batch tile, iterating T=4096 steps with
// h, W_hh, and all small weights in registers.
//
// Transposed formulation: P^T = W_hh x h^T, A (= W_hh, f16) loop-invariant.
// KEY TRICK: A's M-slots are ROW-PERMUTED so the D output layout coincides
// with the next step's B operand layout:
//   A0 M-slot m  <- W_hh row (m<8 ? m   : m+8 )   // rows {0..7, 16..23}
//   A1 M-slot m  <- W_hh row (m<8 ? m+8 : m+16)   // rows {8..15, 24..31}
// Then lane-half hi=0 receives D values for h rows 0..15 (D0: 0..7, D1: 8..15)
// and hi=1 receives rows 16..31 — exactly this code's f16 B order (lanes 0-15 hold
// K=0..15, lanes 16-31 hold K=16..31). Next B = lane-local cvt_pk only;
// NO cross-lane shuffle on the recurrence path.
// Per-step critical path: v_wmma -> v_tanh -> v_cvt_pk -> v_wmma.
__global__ __launch_bounds__(32)
void rnn_elman_wmma(const float* __restrict__ x,        // [B, T]
                    const float* __restrict__ h_state,  // [B, H]
                    const float* __restrict__ W_ih,     // [H]
                    const float* __restrict__ b_ih,     // [H]
                    const float* __restrict__ W_hh,     // [H, H]
                    const float* __restrict__ b_hh,     // [H]
                    const float* __restrict__ W_out,    // [H]
                    const float* __restrict__ b_out,    // [1]
                    float* __restrict__ outs,           // [B, T]
                    float* __restrict__ h_last)         // [B, H]
{
  const int lane = threadIdx.x & 31;
  const int hi   = lane >> 4;      // lane half: 0 or 1
  const int ln   = lane & 15;      // column / M-slot index
  const int b    = blockIdx.x * 16 + ln;   // batch row owned by this lane pair

  // ---- Loop-invariant A tiles in f16, with permuted row assignment.
  //      A layout: lane<16 holds K=0..15 of M-slot ln,
  //      lane>=16 holds K=16..31 of the same M-slot. ----
  const int row0 = (ln < 8) ? ln : (ln + 8);        // A0: rows 0..7,16..23
  const int row1 = (ln < 8) ? (ln + 8) : (ln + 16); // A1: rows 8..15,24..31
  v16h A0, A1;
  const int kb = hi * 16;
  #pragma unroll
  for (int i = 0; i < 4; ++i) {
    const int k0 = kb + 2 * i;
    A0[2*i]     = (_Float16)W_hh[row0 * H_DIM + k0];
    A0[2*i + 1] = (_Float16)W_hh[row0 * H_DIM + k0 + 1];
    A0[8+2*i]   = (_Float16)W_hh[row0 * H_DIM + 8 + k0];
    A0[8+2*i+1] = (_Float16)W_hh[row0 * H_DIM + 8 + k0 + 1];
    A1[2*i]     = (_Float16)W_hh[row1 * H_DIM + k0];
    A1[2*i + 1] = (_Float16)W_hh[row1 * H_DIM + k0 + 1];
    A1[8+2*i]   = (_Float16)W_hh[row1 * H_DIM + 8 + k0];
    A1[8+2*i+1] = (_Float16)W_hh[row1 * H_DIM + 8 + k0 + 1];
  }

  // ---- Per-lane constants in permuted-D layout:
  //      D0[r] <-> h row jA = 16*hi + r ; D1[r] <-> h row jB = 16*hi + 8 + r ----
  float wA[8], wB[8], biasA[8], biasB[8], woA[8], woB[8];
  #pragma unroll
  for (int r = 0; r < 8; ++r) {
    const int jA = 16 * hi + r;
    const int jB = 16 * hi + 8 + r;
    wA[r]    = W_ih[jA];               wB[r]    = W_ih[jB];
    biasA[r] = b_ih[jA] + b_hh[jA];    biasB[r] = b_ih[jB] + b_hh[jB];
    woA[r]   = W_out[jA];              woB[r]   = W_out[jB];
  }
  const float bo = b_out[0];

  // ---- Initial hidden state ----
  float h0[8], h1[8];
  #pragma unroll
  for (int r = 0; r < 8; ++r) {
    h0[r] = h_state[b * H_DIM + 16 * hi + r];
    h1[r] = h_state[b * H_DIM + 16 * hi + 8 + r];
  }

  const float* xp = x    + (size_t)b * T_LEN;
  float*       op = outs + (size_t)b * T_LEN;

  for (int t = 0; t < T_LEN; t += 4) {
    const float4 xv = *(const float4*)(xp + t);
    __builtin_prefetch(xp + t + 64, 0, 1);   // global_prefetch_b8, ~16 iters ahead
    float o4[4];

    #pragma unroll
    for (int s = 0; s < 4; ++s) {
      const float xs = (s == 0) ? xv.x : (s == 1) ? xv.y : (s == 2) ? xv.z : xv.w;

      // ---- B = h^T (f16, K=32 x N=16): pure lane-local packs thanks to the
      //      row-permuted A tiles. VGPR i = K pair {2i,2i+1} of this half's
      //      K-range; h0 covers K offsets 0..7, h1 covers 8..15. ----
      V16HU Bm;
      #pragma unroll
      for (int i = 0; i < 4; ++i) {
        Bm.u[i] = __builtin_bit_cast(
            unsigned, __builtin_amdgcn_cvt_pkrtz(h0[2*i], h0[2*i + 1]));
        Bm.u[4 + i] = __builtin_bit_cast(
            unsigned, __builtin_amdgcn_cvt_pkrtz(h1[2*i], h1[2*i + 1]));
      }

      // ---- C = x_t * W_ih + (b_ih + b_hh)  (off the h critical path) ----
      V8FU C0, C1;
      #pragma unroll
      for (int r = 0; r < 8; ++r) {
        C0.f[r] = fmaf(xs, wA[r], biasA[r]);
        C1.f[r] = fmaf(xs, wB[r], biasB[r]);
      }

      // ---- Two independent WMMAs: D = W_hh(permuted tile) x h^T + C ----
      v8f D0 = __builtin_amdgcn_wmma_f32_16x16x32_f16(
          false, A0, false, Bm.v, (short)0, C0.v, false, false);
      v8f D1 = __builtin_amdgcn_wmma_f32_16x16x32_f16(
          false, A1, false, Bm.v, (short)0, C1.v, false, false);

      // ---- native v_tanh + fused readout partial dot (W_out . h) ----
      float acc = 0.0f;
      #pragma unroll
      for (int r = 0; r < 8; ++r) {
        h0[r] = fast_tanh(D0[r]);
        h1[r] = fast_tanh(D1[r]);
        acc = fmaf(h0[r], woA[r], acc);
        acc = fmaf(h1[r], woB[r], acc);
      }
      acc += __shfl_xor(acc, 16, 32);   // readout reduce: off the h chain
      o4[s] = acc + bo;
    }

    if (hi == 0) {                      // coalesced 16B store per 4 steps
      *(float4*)(op + t) = make_float4(o4[0], o4[1], o4[2], o4[3]);
    }
  }

  // ---- Final hidden state (disjoint (b, j) per lane) ----
  #pragma unroll
  for (int r = 0; r < 8; ++r) {
    h_last[b * H_DIM + 16 * hi + r]     = h0[r];
    h_last[b * H_DIM + 16 * hi + 8 + r] = h1[r];
  }
}

extern "C" void kernel_launch(void* const* d_in, const int* in_sizes, int n_in,
                              void* d_out, int out_size, void* d_ws, size_t ws_size,
                              hipStream_t stream) {
  (void)in_sizes; (void)n_in; (void)out_size; (void)d_ws; (void)ws_size;
  const float* x       = (const float*)d_in[0];  // [B, T, 1]
  const float* h_state = (const float*)d_in[1];  // [1, B, H]
  const float* W_ih    = (const float*)d_in[2];  // [H, 1]
  const float* b_ih    = (const float*)d_in[3];  // [H]
  const float* W_hh    = (const float*)d_in[4];  // [H, H]
  const float* b_hh    = (const float*)d_in[5];  // [H]
  const float* W_out   = (const float*)d_in[6];  // [1, H]
  const float* b_out   = (const float*)d_in[7];  // [1]

  float* outs   = (float*)d_out;                          // [B, T, 1]
  float* h_last = (float*)d_out + (size_t)B_TOT * T_LEN;  // [1, B, H]

  rnn_elman_wmma<<<B_TOT / 16, 32, 0, stream>>>(
      x, h_state, W_ih, b_ih, W_hh, b_hh, W_out, b_out, outs, h_last);
}
